// Predictor_3917010174733
// MI455X (gfx1250) — matrix operand order repaired and checked by reading, not compiled, not run on hardware
//
#include <hip/hip_runtime.h>
#include <hip/hip_bf16.h>

// ---------------------------------------------------------------------------
// GNN message-passing layer for MI455X (gfx1250, wave32, WMMA).
//   edge:  msg = tanh(leaky(edge_inp @ W1e + b1e) @ W2e + b2e)   [E,64]
//   sum_h = segment_sum(msg, dst)                                 [N,64]
//   out   = tanh(leaky(node_inp @ W1n + b1n) @ W2n + b2n)         [N,64]
// MLPs run on v_wmma_f32_16x16x32_f16 (f16 in, f32 accumulate).
// ---------------------------------------------------------------------------

typedef _Float16 v16h __attribute__((ext_vector_type(16)));
typedef _Float16 v8h  __attribute__((ext_vector_type(8)));
typedef float    v8f  __attribute__((ext_vector_type(8)));

constexpr int N_CNT = 100000;
constexpr int E_CNT = 1600000;

// packed-weight half counts
constexpr int PW1E = 3 * 2 * 32 * 16;  // K pad 76->96 (3 chunks), 2 N-tiles
constexpr int PW2E = 4 * 32 * 16;      // K=32, 4 N-tiles
constexpr int PW1N = 5 * 2 * 32 * 16;  // K pad 138->160 (5 chunks), 2 N-tiles
constexpr int PW2N = 4 * 32 * 16;

__device__ __forceinline__ v16h make_frag(v8h lo, v8h hi) {
  v16h r;
#pragma unroll
  for (int i = 0; i < 8; ++i) { r[i] = lo[i]; r[i + 8] = hi[i]; }
  return r;
}

// B fragment from global: lane l holds column N=l%16 of the 32x16 tile (16 K values)
__device__ __forceinline__ v16h load_b_frag(const _Float16* p) {
  return make_frag(*(const v8h*)p, *(const v8h*)(p + 8));
}

// A fragment from LDS row (WMMA 16-bit A swizzle):
// half i<8  -> K = off + i ; half i>=8 -> K = off + 16 + (i-8)
__device__ __forceinline__ v16h load_a_frag(const _Float16* rowBase, int off) {
  return make_frag(*(const v8h*)(rowBase + off), *(const v8h*)(rowBase + off + 16));
}

__device__ __forceinline__ v8f bias_frag(float bv) {
  v8f c;
#pragma unroll
  for (int i = 0; i < 8; ++i) c[i] = bv;
  return c;
}

// ---------------------------------------------------------------------------
// Pack f32 weights -> f16 WMMA B-layout:  [(kchunk,ntile)][lane: col=lane%16][16 K values]
// ---------------------------------------------------------------------------
__global__ void pack_kernel(const float* __restrict__ W1e, const float* __restrict__ W2e,
                            const float* __restrict__ W1n, const float* __restrict__ W2n,
                            _Float16* __restrict__ pw1e, _Float16* __restrict__ pw2e,
                            _Float16* __restrict__ pw1n, _Float16* __restrict__ pw2n) {
  const int t = threadIdx.x + blockIdx.x * blockDim.x;
  const int stride = blockDim.x * gridDim.x;
  for (int idx = t; idx < PW1E; idx += stride) {
    int i = idx & 15, lane = (idx >> 4) & 31, nt = (idx >> 9) & 1, c = idx >> 10;
    int k = c * 32 + ((i & 8) << 1) + (lane >> 4) * 8 + (i & 7), col = nt * 16 + (lane & 15);
    pw1e[idx] = (k < 76) ? (_Float16)W1e[k * 32 + col] : (_Float16)0.f;
  }
  for (int idx = t; idx < PW2E; idx += stride) {
    int i = idx & 15, lane = (idx >> 4) & 31, nt = (idx >> 9) & 3;
    pw2e[idx] = (_Float16)W2e[(((i & 8) << 1) + (lane >> 4) * 8 + (i & 7)) * 64 + nt * 16 + (lane & 15)];
  }
  for (int idx = t; idx < PW1N; idx += stride) {
    int i = idx & 15, lane = (idx >> 4) & 31, nt = (idx >> 9) & 1, c = idx >> 10;
    int k = c * 32 + ((i & 8) << 1) + (lane >> 4) * 8 + (i & 7), col = nt * 16 + (lane & 15);
    pw1n[idx] = (k < 138) ? (_Float16)W1n[k * 32 + col] : (_Float16)0.f;
  }
  for (int idx = t; idx < PW2N; idx += stride) {
    int i = idx & 15, lane = (idx >> 4) & 31, nt = (idx >> 9) & 3;
    pw2n[idx] = (_Float16)W2n[(((i & 8) << 1) + (lane >> 4) * 8 + (i & 7)) * 64 + nt * 16 + (lane & 15)];
  }
}

__global__ void zero_kernel(float4* __restrict__ p, int n4) {
  int i = threadIdx.x + blockIdx.x * blockDim.x;
  if (i < n4) p[i] = float4{0.f, 0.f, 0.f, 0.f};
}

// ---------------------------------------------------------------------------
// Edge kernel: 8 waves/block, one 16-edge tile per wave. E % 128 == 0.
// edge_inp = [pos[src](2) pos[dst](2) u[src](8) h[src](64)] = 76, pad K->96.
// ---------------------------------------------------------------------------
__global__ __launch_bounds__(256) void edge_kernel(
    const float* __restrict__ h, const float* __restrict__ u,
    const float* __restrict__ pos, const int* __restrict__ src,
    const int* __restrict__ dst,
    const _Float16* __restrict__ pw1e, const float* __restrict__ b1e,
    const _Float16* __restrict__ pw2e, const float* __restrict__ b2e,
    float* __restrict__ sumh) {
  constexpr int INS = 104;  // row stride in halves (96 used, 16B-aligned pad)
  constexpr int YS = 40;    // y row stride in halves (32 used)
  __shared__ __align__(16) _Float16 sIn[8][16 * INS];
  __shared__ __align__(16) _Float16 sY[8][16 * YS];
  __shared__ int sDst[8][16];

  const int wave = threadIdx.x >> 5;
  const int lane = threadIdx.x & 31;
  const int m = lane & 15;   // row within tile
  const int p = lane >> 4;   // half-wave
  const int e = (blockIdx.x * 8 + wave) * 16 + m;

  // ---- stage edge_inp tile into LDS as f16 ----
  _Float16* row = &sIn[wave][m * INS];
  const int s = src[e];
  const float4* h4 = (const float4*)(h + (long)s * 64);
  if (p == 0) {
    const int d = dst[e];
    sDst[wave][m] = d;
    row[0] = (_Float16)pos[2 * s];
    row[1] = (_Float16)pos[2 * s + 1];
    row[2] = (_Float16)pos[2 * d];
    row[3] = (_Float16)pos[2 * d + 1];
#pragma unroll
    for (int j = 0; j < 8; ++j) row[4 + j] = (_Float16)u[8 * s + j];
#pragma unroll
    for (int j = 0; j < 8; ++j) {
      float4 v = h4[j];
      row[12 + 4 * j + 0] = (_Float16)v.x; row[12 + 4 * j + 1] = (_Float16)v.y;
      row[12 + 4 * j + 2] = (_Float16)v.z; row[12 + 4 * j + 3] = (_Float16)v.w;
    }
  } else {
#pragma unroll
    for (int j = 8; j < 16; ++j) {
      float4 v = h4[j];
      row[12 + 4 * j + 0] = (_Float16)v.x; row[12 + 4 * j + 1] = (_Float16)v.y;
      row[12 + 4 * j + 2] = (_Float16)v.z; row[12 + 4 * j + 3] = (_Float16)v.w;
    }
#pragma unroll
    for (int k = 76; k < 96; ++k) row[k] = (_Float16)0.f;
  }
  __syncthreads();

  // ---- GEMM1: [16x96] x [96x32], bias as C operand ----
  v8f acc1[2];
#pragma unroll
  for (int nt = 0; nt < 2; ++nt) acc1[nt] = bias_frag(b1e[nt * 16 + m]);
#pragma unroll
  for (int c = 0; c < 3; ++c) {
    v16h a = load_a_frag(row, c * 32 + p * 8);
#pragma unroll
    for (int nt = 0; nt < 2; ++nt) {
      v16h b = load_b_frag(pw1e + ((c * 2 + nt) * 32 + lane) * 16);
      acc1[nt] = __builtin_amdgcn_wmma_f32_16x16x32_f16(false, a, false, b,
                                                        (short)0, acc1[nt], false, false);
    }
  }

  // ---- LeakyReLU, restage y as f16 A-operand ----
#pragma unroll
  for (int nt = 0; nt < 2; ++nt)
#pragma unroll
    for (int r = 0; r < 8; ++r) {
      float v = acc1[nt][r];
      v = (v >= 0.f) ? v : 0.01f * v;
      sY[wave][(r + p * 8) * YS + nt * 16 + m] = (_Float16)v;
    }
  __syncthreads();

  // ---- GEMM2: [16x32] x [32x64], tanh, atomic scatter into sum_h[dst] ----
  v16h ay = load_a_frag(&sY[wave][m * YS], p * 8);
  v8f acc2[4];
#pragma unroll
  for (int nt = 0; nt < 4; ++nt) {
    v16h b = load_b_frag(pw2e + (nt * 32 + lane) * 16);
    acc2[nt] = __builtin_amdgcn_wmma_f32_16x16x32_f16(false, ay, false, b,
                                                      (short)0, bias_frag(b2e[nt * 16 + m]),
                                                      false, false);
  }
#pragma unroll
  for (int r = 0; r < 8; ++r) {
    const int d = sDst[wave][r + p * 8];
    float* base = sumh + (long)d * 64;
#pragma unroll
    for (int nt = 0; nt < 4; ++nt) atomicAdd(base + nt * 16 + m, tanhf(acc2[nt][r]));
  }
}

// ---------------------------------------------------------------------------
// Node kernel: node_inp = [pos(2) h(64) sum_h(64) u(8)] = 138, pad K->160.
// N = 6250 full tiles; waves past 6250 clamp loads and skip stores.
// ---------------------------------------------------------------------------
__global__ __launch_bounds__(256) void node_kernel(
    const float* __restrict__ h, const float* __restrict__ u,
    const float* __restrict__ pos, const float* __restrict__ sumh,
    const _Float16* __restrict__ pw1n, const float* __restrict__ b1n,
    const _Float16* __restrict__ pw2n, const float* __restrict__ b2n,
    float* __restrict__ out) {
  constexpr int INS = 168;  // row stride in halves (160 used)
  constexpr int YS = 40;
  __shared__ __align__(16) _Float16 sIn[8][16 * INS];
  __shared__ __align__(16) _Float16 sY[8][16 * YS];

  const int wave = threadIdx.x >> 5;
  const int lane = threadIdx.x & 31;
  const int m = lane & 15;
  const int p = lane >> 4;
  const int tile = blockIdx.x * 8 + wave;
  const bool valid = tile < (N_CNT / 16);
  const int tc = valid ? tile : (N_CNT / 16) - 1;
  const int nb = tc * 16;
  const int node = nb + m;  // < N by construction

  _Float16* row = &sIn[wave][m * INS];
  if (p == 0) {
    row[0] = (_Float16)pos[2 * node];
    row[1] = (_Float16)pos[2 * node + 1];
    const float4* h4 = (const float4*)(h + (long)node * 64);
#pragma unroll
    for (int j = 0; j < 16; ++j) {
      float4 v = h4[j];
      row[2 + 4 * j + 0] = (_Float16)v.x; row[2 + 4 * j + 1] = (_Float16)v.y;
      row[2 + 4 * j + 2] = (_Float16)v.z; row[2 + 4 * j + 3] = (_Float16)v.w;
    }
  } else {
    const float4* s4 = (const float4*)(sumh + (long)node * 64);
#pragma unroll
    for (int j = 0; j < 16; ++j) {
      float4 v = s4[j];
      row[66 + 4 * j + 0] = (_Float16)v.x; row[66 + 4 * j + 1] = (_Float16)v.y;
      row[66 + 4 * j + 2] = (_Float16)v.z; row[66 + 4 * j + 3] = (_Float16)v.w;
    }
#pragma unroll
    for (int j = 0; j < 8; ++j) row[130 + j] = (_Float16)u[8 * node + j];
#pragma unroll
    for (int k = 138; k < 160; ++k) row[k] = (_Float16)0.f;
  }
  __syncthreads();

  v8f acc1[2];
#pragma unroll
  for (int nt = 0; nt < 2; ++nt) acc1[nt] = bias_frag(b1n[nt * 16 + m]);
#pragma unroll
  for (int c = 0; c < 5; ++c) {
    v16h a = load_a_frag(row, c * 32 + p * 8);
#pragma unroll
    for (int nt = 0; nt < 2; ++nt) {
      v16h b = load_b_frag(pw1n + ((c * 2 + nt) * 32 + lane) * 16);
      acc1[nt] = __builtin_amdgcn_wmma_f32_16x16x32_f16(false, a, false, b,
                                                        (short)0, acc1[nt], false, false);
    }
  }
#pragma unroll
  for (int nt = 0; nt < 2; ++nt)
#pragma unroll
    for (int r = 0; r < 8; ++r) {
      float v = acc1[nt][r];
      v = (v >= 0.f) ? v : 0.01f * v;
      sY[wave][(r + p * 8) * YS + nt * 16 + m] = (_Float16)v;
    }
  __syncthreads();

  v16h ay = load_a_frag(&sY[wave][m * YS], p * 8);
  v8f acc2[4];
#pragma unroll
  for (int nt = 0; nt < 4; ++nt) {
    v16h b = load_b_frag(pw2n + (nt * 32 + lane) * 16);
    acc2[nt] = __builtin_amdgcn_wmma_f32_16x16x32_f16(false, ay, false, b,
                                                      (short)0, bias_frag(b2n[nt * 16 + m]),
                                                      false, false);
  }
  if (valid) {
#pragma unroll
    for (int r = 0; r < 8; ++r) {
      float* base = out + (long)(nb + r + p * 8) * 64;
#pragma unroll
      for (int nt = 0; nt < 4; ++nt) base[nt * 16 + m] = tanhf(acc2[nt][r]);
    }
  }
}

// ---------------------------------------------------------------------------
extern "C" void kernel_launch(void* const* d_in, const int* in_sizes, int n_in,
                              void* d_out, int out_size, void* d_ws, size_t ws_size,
                              hipStream_t stream) {
  const float* h   = (const float*)d_in[0];
  const float* u   = (const float*)d_in[1];
  const float* pos = (const float*)d_in[2];
  const int* src   = (const int*)d_in[3];
  const int* dst   = (const int*)d_in[4];
  const float* W1e = (const float*)d_in[5];
  const float* b1e = (const float*)d_in[6];
  const float* W2e = (const float*)d_in[7];
  const float* b2e = (const float*)d_in[8];
  const float* W1n = (const float*)d_in[9];
  const float* b1n = (const float*)d_in[10];
  const float* W2n = (const float*)d_in[11];
  const float* b2n = (const float*)d_in[12];
  float* out = (float*)d_out;

  float* sumh = (float*)d_ws;                                   // N*64 f32
  _Float16* pk = (_Float16*)((char*)d_ws + (size_t)N_CNT * 64 * sizeof(float));
  _Float16* pw1e = pk;
  _Float16* pw2e = pk + PW1E;
  _Float16* pw1n = pk + PW1E + PW2E;
  _Float16* pw2n = pk + PW1E + PW2E + PW1N;

  pack_kernel<<<12, 256, 0, stream>>>(W1e, W2e, W1n, W2n, pw1e, pw2e, pw1n, pw2n);

  const int n4 = N_CNT * 64 / 4;
  zero_kernel<<<(n4 + 255) / 256, 256, 0, stream>>>((float4*)sumh, n4);

  edge_kernel<<<E_CNT / 128, 256, 0, stream>>>(h, u, pos, src, dst,
                                               pw1e, b1e, pw2e, b2e, sumh);

  const int ntiles = N_CNT / 16;                 // 6250
  node_kernel<<<(ntiles + 7) / 8, 256, 0, stream>>>(h, u, pos, sumh,
                                                    pw1n, b1n, pw2n, b2n, out);
}
